// gcn_40003325395285
// MI455X (gfx1250) — matrix operand order repaired and checked by reading, not compiled, not run on hardware
//
#include <hip/hip_runtime.h>

typedef __attribute__((ext_vector_type(16))) _Float16 v16h;
typedef __attribute__((ext_vector_type(8)))  float    v8f;

union v16h_u { v16h v; unsigned u[8]; };
union pk2    { _Float16 h[2]; unsigned u; };

#define NB   64
#define TT   512
#define VV   25
#define CIN  3
#define CH   64
#define FF   25
#define GG   100            // 4*F
#define PP   (NB*VV)        // 1600 pixels
#define GPAD 112            // 7 column tiles of 16

__device__ __forceinline__ float hsig(float x) {
    return fminf(fmaxf(0.2f*x + 0.5f, 0.0f), 1.0f);
}

// ISA 7.12.2: 16-bit A-matrix 16x32 fragment. Pair j (0..7), half = lane>>4.
__device__ __forceinline__ int a_k(int j, int half) {
    return (j < 4 ? 2*j : 16 + 2*(j-4)) + 8*half;   // always even
}
// 16-bit B-matrix 32x16 fragment: pair j holds same K as a_k(j, half) (even).
__device__ __forceinline__ int b_k(int j, int half) {
    return (j < 4 ? 2*j : 16 + 2*(j-4)) + 8*half;
}

// -------------------------------------------------------------------------
// Kernel 1: zx[t][p][g] = relu(x[n,t,v,:]Wc + bc) @ W_lstm + b_lstm
// W_lstm staged in LDS pre-packed as f16 pairs in B-fragment order and
// zero-padded to 112 cols -> unconditional ds_load_b32 fragment builds.
// -------------------------------------------------------------------------
__global__ __launch_bounds__(256) void k_conv_zx(
    const float* __restrict__ x, const float* __restrict__ Wc,
    const float* __restrict__ bc, const float* __restrict__ Wl,
    const float* __restrict__ bl, float* __restrict__ zx)
{
    __shared__ float    sWc[CIN*CH];
    __shared__ float    sbc[CH];
    __shared__ float    sbl[GG];
    __shared__ unsigned sWlp[32*GPAD];   // [k/2][col], f16 pair {k, k+1}
    const int tid = threadIdx.x;
    for (int i = tid; i < CIN*CH; i += 256) sWc[i] = Wc[i];
    for (int i = tid; i < CH;     i += 256) sbc[i] = bc[i];
    for (int i = tid; i < GG;     i += 256) sbl[i] = bl[i];
    for (int i = tid; i < 32*GPAD; i += 256) {
        const int kh = i / GPAD, col = i % GPAD;
        pk2 p;
        p.h[0] = (_Float16)((col < GG) ? Wl[(2*kh  )*GG + col] : 0.f);
        p.h[1] = (_Float16)((col < GG) ? Wl[(2*kh+1)*GG + col] : 0.f);
        sWlp[i] = p.u;
    }
    __syncthreads();

    const int wave = tid >> 5;
    const int lane = tid & 31;
    const int m    = lane & 15;
    const int half = lane >> 4;
    const int tile  = blockIdx.x * 8 + wave;   // 51200 tiles total
    const int t     = tile / 100;
    const int ptile = tile % 100;
    const int p = ptile * 16 + m;
    const int n = p / VV, v = p % VV;
    const float* xr = x + (((size_t)n*TT + t)*VV + v)*CIN;
    const float x0 = xr[0], x1v = xr[1], x2 = xr[2];

    v8f acc[7];
    #pragma unroll
    for (int i = 0; i < 7; ++i) acc[i] = {};

    #pragma unroll
    for (int kk = 0; kk < 2; ++kk) {
        v16h a;
        #pragma unroll
        for (int j = 0; j < 8; ++j) {            // conv on the fly -> f16 A frag
            const int c0 = kk*32 + a_k(j, half);
            const float f0 = fmaxf(x0*sWc[c0]   + x1v*sWc[CH+c0]   + x2*sWc[2*CH+c0]   + sbc[c0],   0.f);
            const float f1 = fmaxf(x0*sWc[c0+1] + x1v*sWc[CH+c0+1] + x2*sWc[2*CH+c0+1] + sbc[c0+1], 0.f);
            a[2*j]   = (_Float16)f0;
            a[2*j+1] = (_Float16)f1;
        }
        #pragma unroll
        for (int nt = 0; nt < 7; ++nt) {
            const int col = nt*16 + m;
            v16h_u b;
            #pragma unroll
            for (int j = 0; j < 8; ++j)          // unconditional packed loads
                b.u[j] = sWlp[(kk*16 + (b_k(j, half) >> 1))*GPAD + col];
            acc[nt] = __builtin_amdgcn_wmma_f32_16x16x32_f16(
                false, a, false, b.v, (short)0, acc[nt], false, false);
        }
    }
    #pragma unroll
    for (int nt = 0; nt < 7; ++nt) {
        const int col = nt*16 + m;
        if (col < GG) {
            #pragma unroll
            for (int r = 0; r < 8; ++r) {
                const int p2 = ptile*16 + r + 8*half;
                zx[((size_t)t*PP + p2)*GG + col] = acc[nt][r] + sbl[col];
            }
        }
    }
}

// -------------------------------------------------------------------------
// Kernel 2: per-pixel LSTM. 1 wave owns 16 independent pixels. h lives in
// LDS as zero-padded f16 (written b16 by the gate pass, read back as packed
// b32 pairs for the A fragment). U fragments are VGPR-resident. Next
// timestep's zx rows are prefetched (global_prefetch_b8) to hide latency
// on the serialized recurrence.
// -------------------------------------------------------------------------
__global__ __launch_bounds__(32) void k_lstm(
    const float* __restrict__ zx, const float* __restrict__ U,
    float* __restrict__ hsb)
{
    __shared__ _Float16 sh[16*32];      // h, zero-padded K 25->32
    __shared__ float    sc[16*FF];      // cell state
    __shared__ float    sg[16*GPAD];    // WMMA gate output staging
    const int lane = threadIdx.x;
    const int m    = lane & 15;
    const int half = lane >> 4;
    const int p0   = blockIdx.x * 16;

    for (int i = lane; i < 16*32; i += 32) sh[i] = (_Float16)0.f;
    for (int i = lane; i < 16*FF; i += 32) sc[i] = 0.f;

    v16h bU[7];                         // U (25x100) fragments in VGPRs
    #pragma unroll
    for (int nt = 0; nt < 7; ++nt) {
        const int col = nt*16 + m;
        const int colc = col < GG ? col : 0;
        #pragma unroll
        for (int j = 0; j < 8; ++j) {
            const int k0 = b_k(j, half);
            // clamped addresses + select: v_cndmask, no exec branching
            float f0 = U[(k0   < FF ? k0   : 0)*GG + colc];
            float f1 = U[(k0+1 < FF ? k0+1 : 0)*GG + colc];
            f0 = (k0   < FF && col < GG) ? f0 : 0.f;
            f1 = (k0+1 < FF && col < GG) ? f1 : 0.f;
            bU[nt][2*j]   = (_Float16)f0;
            bU[nt][2*j+1] = (_Float16)f1;
        }
    }
    __syncthreads();

    const unsigned* shp = (const unsigned*)sh;
    for (int t = 0; t < TT; ++t) {
        // prefetch next step's z rows (16 x 100 f32 = 6400 B, 200 B / lane)
        if (t + 1 < TT)
            __builtin_prefetch((const char*)(zx + ((size_t)(t+1)*PP + p0)*GG)
                               + lane*200, 0, 1);
        v16h_u a;
        #pragma unroll
        for (int j = 0; j < 8; ++j)
            a.u[j] = shp[m*16 + (a_k(j, half) >> 1)];
        #pragma unroll
        for (int nt = 0; nt < 7; ++nt) {
            v8f acc = {};
            acc = __builtin_amdgcn_wmma_f32_16x16x32_f16(
                false, a.v, false, bU[nt], (short)0, acc, false, false);
            const int col = nt*16 + m;
            #pragma unroll
            for (int r = 0; r < 8; ++r)
                sg[(r + 8*half)*GPAD + col] = acc[r];
        }
        __syncthreads();
        for (int q = lane; q < 16*FF; q += 32) {
            const int row = q / FF, f = q % FF;
            const float* z = zx + ((size_t)t*PP + p0 + row)*GG;
            const float gi = sg[row*GPAD + f]         + z[f];
            const float gf = sg[row*GPAD + FF + f]    + z[FF + f];
            const float gg = sg[row*GPAD + 2*FF + f]  + z[2*FF + f];
            const float go = sg[row*GPAD + 3*FF + f]  + z[3*FF + f];
            const float cc = hsig(gf)*sc[row*FF + f] + hsig(gi)*tanhf(gg);
            const float hh = hsig(go)*tanhf(cc);
            sc[row*FF + f] = cc;
            sh[row*32 + f] = (_Float16)hh;      // ds_store_b16
            const int pr = p0 + row;
            hsb[(((size_t)(pr / VV)*TT + t)*VV + (pr % VV))*FF + f] = hh;
        }
        __syncthreads();
    }
}

// -------------------------------------------------------------------------
// Kernel 3: attention + aggregation per (n,t). coefs and x1 are produced
// directly as zero-padded f16 in LDS so both WMMA fragments are packed
// unconditional b32 loads. out = coefs(25x25) @ x1(25x64), 2x4 tiles.
// -------------------------------------------------------------------------
__global__ __launch_bounds__(64) void k_attn_agg(
    const float* __restrict__ x, const float* __restrict__ Wc,
    const float* __restrict__ bc, const float* __restrict__ bias,
    const float* __restrict__ hsb, float* __restrict__ out)
{
    __shared__ _Float16 sA[32*32];      // coefs [row][k], zero-padded
    __shared__ _Float16 sBt[CH*32];     // x1 transposed [col][k], zero-padded
    __shared__ float    sWc[CIN*CH];
    __shared__ float    sbc[CH];
    __shared__ float    sBias[VV*VV];
    const int tid = threadIdx.x;
    const int n = blockIdx.x / TT, t = blockIdx.x % TT;

    for (int i = tid; i < 32*32;  i += 64) sA[i]  = (_Float16)0.f;
    for (int i = tid; i < CH*32;  i += 64) sBt[i] = (_Float16)0.f;
    for (int i = tid; i < CIN*CH; i += 64) sWc[i] = Wc[i];
    for (int i = tid; i < CH;     i += 64) sbc[i] = bc[i];
    for (int i = tid; i < VV*VV;  i += 64) sBias[i] = bias[i];
    __syncthreads();

    {   // x1 tile (25x64): thread owns one output channel column
        const int col = tid;
        const float w0 = sWc[col], w1 = sWc[CH+col], w2 = sWc[2*CH+col];
        const float bb = sbc[col];
        #pragma unroll
        for (int v = 0; v < VV; ++v) {
            const float* xr = x + (((size_t)n*TT + t)*VV + v)*CIN;
            sBt[col*32 + v] =
                (_Float16)fmaxf(xr[0]*w0 + xr[1]*w1 + xr[2]*w2 + bb, 0.f);
        }
    }
    if (tid < VV) {   // softmax row (25 wide) per thread, fully unrolled
        const int row = tid;
        const float* fr = hsb + (((size_t)n*TT + t)*VV + row)*FF;
        float buf[VV];
        float mx = -1e30f;
        #pragma unroll
        for (int w = 0; w < VV; ++w) {
            const float xv = fr[w];
            const float s = (xv > 0.f ? xv : 0.2f*xv) + sBias[row*VV + w];
            buf[w] = s;
            mx = fmaxf(mx, s);
        }
        float sum = 0.f;
        #pragma unroll
        for (int w = 0; w < VV; ++w) { buf[w] = expf(buf[w]-mx); sum += buf[w]; }
        const float inv = 1.f / sum;
        #pragma unroll
        for (int w = 0; w < VV; ++w) sA[row*32 + w] = (_Float16)(buf[w]*inv);
    }
    __syncthreads();

    const int wave = tid >> 5, lane = tid & 31;
    const int m = lane & 15, half = lane >> 4;
    const unsigned* sAp  = (const unsigned*)sA;
    const unsigned* sBtp = (const unsigned*)sBt;
    v16h_u a;
    #pragma unroll
    for (int j = 0; j < 8; ++j)
        a.u[j] = sAp[(wave*16 + m)*16 + (a_k(j, half) >> 1)];
    #pragma unroll
    for (int ntile = 0; ntile < 4; ++ntile) {
        const int col = ntile*16 + m;
        v16h_u b;
        #pragma unroll
        for (int j = 0; j < 8; ++j)
            b.u[j] = sBtp[col*16 + (b_k(j, half) >> 1)];
        v8f acc = {};
        acc = __builtin_amdgcn_wmma_f32_16x16x32_f16(
            false, a.v, false, b.v, (short)0, acc, false, false);
        #pragma unroll
        for (int r = 0; r < 8; ++r) {
            const int vrow = wave*16 + r + 8*half;
            if (vrow < VV)
                out[(((size_t)n*TT + t)*VV + vrow)*CH + col] = acc[r];
        }
    }
}

extern "C" void kernel_launch(void* const* d_in, const int* in_sizes, int n_in,
                              void* d_out, int out_size, void* d_ws, size_t ws_size,
                              hipStream_t stream) {
    (void)in_sizes; (void)n_in; (void)out_size; (void)ws_size;
    const float* x    = (const float*)d_in[0];
    const float* Wc   = (const float*)d_in[1];
    const float* bc   = (const float*)d_in[2];
    const float* Wl   = (const float*)d_in[3];
    const float* U    = (const float*)d_in[4];
    const float* bl   = (const float*)d_in[5];
    const float* bias = (const float*)d_in[6];
    float* out = (float*)d_out;

    // workspace: zx (T*P*100 f32 = 327.7 MB) then hs (N*T*V*F f32 = 81.9 MB)
    float* zx  = (float*)d_ws;
    float* hsb = zx + (size_t)TT*PP*GG;

    // 51200 M-tiles, 8 waves/block
    k_conv_zx<<<dim3(6400), dim3(256), 0, stream>>>(x, Wc, bc, Wl, bl, zx);
    // 100 independent 16-pixel LSTM blocks (sequential over T inside)
    k_lstm<<<dim3(PP/16), dim3(32), 0, stream>>>(zx, U, hsb);
    // one block per (n,t)
    k_attn_agg<<<dim3(NB*TT), dim3(64), 0, stream>>>(x, Wc, bc, bias, hsb, out);
}
